// BiGCNNet_9354438770820
// MI455X (gfx1250) — compile-verified, matrix operand order checked by reading, not run on hardware
//
#include <hip/hip_runtime.h>
#include <hip/hip_bf16.h>

typedef __attribute__((ext_vector_type(16))) __bf16 v16bf;
typedef __attribute__((ext_vector_type(8)))  __bf16 v8bf;
typedef __attribute__((ext_vector_type(8)))  float  v8f;

#define TPB 256

static __device__ __forceinline__ void atomic_add_f32(float* p, float v) {
    __hip_atomic_fetch_add(p, v, __ATOMIC_RELAXED, __HIP_MEMORY_SCOPE_AGENT);
}

// ---------- data prep ----------
__global__ void k_cast_x0(const float* __restrict__ x, const float* __restrict__ xda,
                          __bf16* __restrict__ x0b, int N) {
    int idx = blockIdx.x * blockDim.x + threadIdx.x;
    if (idx >= N * 128) return;
    int i = idx >> 7, c = idx & 127;
    float v = (c < 96) ? x[i * 96 + c] : xda[i * 32 + (c - 96)];
    x0b[idx] = (__bf16)v;
}

__global__ void k_fill(float* __restrict__ p, float v, int n) {
    int i = blockIdx.x * blockDim.x + threadIdx.x;
    if (i < n) p[i] = v;
}

__global__ void k_count_deg(const long long* __restrict__ e0, const long long* __restrict__ e1,
                            float* __restrict__ degTD, float* __restrict__ degBU, int E) {
    int i = blockIdx.x * blockDim.x + threadIdx.x;
    if (i >= E) return;
    atomic_add_f32(&degTD[(int)e1[i]], 1.0f);   // TD: dst = edge[1]
    atomic_add_f32(&degBU[(int)e0[i]], 1.0f);   // BU: dst = edge[0]
}

__global__ void k_rsqrt(float* __restrict__ p, int n) {
    int i = blockIdx.x * blockDim.x + threadIdx.x;
    if (i < n) p[i] = rsqrtf(p[i]);
}

// Pack W[128,128] (f32, row-major, k-major) into WMMA B-fragment order:
// frag index = ((wv*4 + ks)*32 + lane)*16 + e  -> 16 contiguous bf16 per lane.
__global__ void k_pack_w(const float* __restrict__ w, __bf16* __restrict__ wf) {
    int idx = blockIdx.x * blockDim.x + threadIdx.x;   // 16384
    if (idx >= 128 * 128) return;
    int e    = idx & 15;
    int lane = (idx >> 4) & 31;
    int ks   = (idx >> 9) & 3;
    int wv   = idx >> 11;
    int half = lane >> 4, lm = lane & 15;
    int v = e >> 1, odd = e & 1;
    int kb = ((v < 4) ? (2 * v + 8 * half) : (16 + 2 * (v - 4) + 8 * half)) + odd;
    int k   = ks * 32 + kb;
    int col = wv * 16 + lm;
    wf[idx] = (__bf16)w[k * 128 + col];
}

// ---------- WMMA GEMM: C[M,128] = A[M,128](bf16) @ W(pre-packed bf16 frags) ----------
// grid.x = M/16 row tiles; 8 waves per block, wave wv -> column tile wv*16.
// A strip staged through LDS (272B row stride: 16B-aligned, 4-bank skew per row).
__global__ void k_gemm_wmma_bf16(const __bf16* __restrict__ A,
                                 const __bf16* __restrict__ Bf,
                                 float* __restrict__ C) {
    __shared__ __bf16 As[16][136];
    int tid = threadIdx.x;
    {   // cooperative stage of the 16x128 A strip (one b128 load/store per thread)
        int r = tid >> 4, cst = (tid & 15) * 8;
        *(v8bf*)&As[r][cst] =
            *(const v8bf*)(A + (size_t)blockIdx.x * 2048 + r * 128 + cst);
    }
    __syncthreads();
    int wv = tid >> 5, lane = tid & 31, half = lane >> 4, lm = lane & 15;
    v8f acc = {};
#pragma unroll
    for (int ks = 0; ks < 4; ++ks) {
        int k0 = ks * 32;
        v8bf a0 = *(const v8bf*)&As[lm][k0 + 8 * half];        // K = k0+8h .. +7
        v8bf a1 = *(const v8bf*)&As[lm][k0 + 16 + 8 * half];   // K = k0+16+8h .. +7
        v16bf a = __builtin_shufflevector(a0, a1, 0, 1, 2, 3, 4, 5, 6, 7,
                                                  8, 9, 10, 11, 12, 13, 14, 15);
        v16bf b = *(const v16bf*)(Bf + ((((wv << 2) | ks) << 5 | lane) << 4));
        acc = __builtin_amdgcn_wmma_f32_16x16x32_bf16(false, a, false, b,
                                                      (short)0, acc, false, false);
    }
    int col = wv * 16 + lm;
    int rbase = blockIdx.x * 16 + 8 * half;
#pragma unroll
    for (int r = 0; r < 8; ++r)
        C[(size_t)(rbase + r) * 128 + col] = acc[r];
}

// ---------- GCN aggregation ----------
// out[i,c] = hw[i,c]*dinv[i]^2 + bias[c]   (self loop + bias, also zero-inits)
__global__ void k_conv_init(const float* __restrict__ hw, const float* __restrict__ dinv,
                            const float* __restrict__ bias, float* __restrict__ out, int N) {
    int idx = blockIdx.x * blockDim.x + threadIdx.x;   // over N*32 float4s
    if (idx >= N * 32) return;
    int i = idx >> 5, c4 = (idx & 31) * 4;
    float di = dinv[i];
    float dd = di * di;
    float4 h = *(const float4*)(hw + (size_t)i * 128 + c4);
    float4 b = *(const float4*)(bias + c4);
    float4 o = { h.x * dd + b.x, h.y * dd + b.y, h.z * dd + b.z, h.w * dd + b.w };
    *(float4*)(out + (size_t)i * 128 + c4) = o;
}

// one wave per edge: out[dst] += hw[src] * dinv[src]*dinv[dst]
__global__ void k_conv_edge(const float* __restrict__ hw,
                            const long long* __restrict__ srcA,
                            const long long* __restrict__ dstA,
                            const float* __restrict__ dinv,
                            float* __restrict__ out, int E) {
    int wid  = (blockIdx.x * blockDim.x + threadIdx.x) >> 5;
    int lane = threadIdx.x & 31;
    if (wid >= E) return;
    int s = (int)srcA[wid], d = (int)dstA[wid];
    float nrm = dinv[s] * dinv[d];
    const float4 v = *(const float4*)(hw + (size_t)s * 128 + lane * 4);
    float* o = out + (size_t)d * 128 + lane * 4;
    atomic_add_f32(o + 0, v.x * nrm);
    atomic_add_f32(o + 1, v.y * nrm);
    atomic_add_f32(o + 2, v.z * nrm);
    atomic_add_f32(o + 3, v.w * nrm);
}

__global__ void k_relu_cast(const float* __restrict__ h, __bf16* __restrict__ hb, int n4) {
    int i = blockIdx.x * blockDim.x + threadIdx.x;   // over n/4
    if (i >= n4) return;
    float4 v = *(const float4*)(h + (size_t)i * 4);
    __bf16 r[4];
    r[0] = (__bf16)(v.x > 0.f ? v.x : 0.f);
    r[1] = (__bf16)(v.y > 0.f ? v.y : 0.f);
    r[2] = (__bf16)(v.z > 0.f ? v.z : 0.f);
    r[3] = (__bf16)(v.w > 0.f ? v.w : 0.f);
    *(uint2*)(hb + (size_t)i * 4) = *(uint2*)r;
}

__global__ void k_gather_root(const float* __restrict__ h1, const long long* __restrict__ ri,
                              float* __restrict__ rx, int B) {
    int idx = blockIdx.x * blockDim.x + threadIdx.x;
    if (idx >= B * 128) return;
    int b = idx >> 7, c = idx & 127;
    rx[idx] = h1[(size_t)((int)ri[b]) * 128 + c];
}

// rt[b,o] = sum_k relu(x0[root_b,k]) * W2[128+k, o]   (per-graph constant of layer 2)
__global__ void k_root_term(const float* __restrict__ x, const float* __restrict__ xda,
                            const long long* __restrict__ ri,
                            const float* __restrict__ w2, float* __restrict__ rt) {
    int b = blockIdx.x, o = threadIdx.x;   // 128 threads
    int rb = (int)ri[b];
    float acc = 0.f;
    for (int k = 0; k < 128; ++k) {
        float xr = (k < 96) ? x[rb * 96 + k] : xda[rb * 32 + (k - 96)];
        xr = xr > 0.f ? xr : 0.f;
        acc += xr * w2[(128 + k) * 128 + o];
    }
    rt[b * 128 + o] = acc;
}

__global__ void k_add_rt(float* __restrict__ hw2, const float* __restrict__ rt,
                         const long long* __restrict__ batch, int N) {
    int idx = blockIdx.x * blockDim.x + threadIdx.x;   // over N*32 float4s
    if (idx >= N * 32) return;
    int i = idx >> 5, c4 = (idx & 31) * 4;
    int g = (int)batch[i];
    float4 a = *(const float4*)(hw2 + (size_t)i * 128 + c4);
    float4 r = *(const float4*)(rt + (size_t)g * 128 + c4);
    float4 o = { a.x + r.x, a.y + r.y, a.z + r.z, a.w + r.w };
    *(float4*)(hw2 + (size_t)i * 128 + c4) = o;
}

// segment-mean of relu(out2) over contiguous graphs + root2 concat
__global__ void k_readout(const float* __restrict__ out2, const float* __restrict__ rootx2,
                          const long long* __restrict__ ri, float* __restrict__ g,
                          int goff, int npg) {
    int b = blockIdx.x, c = threadIdx.x;   // 128 threads
    int start = (int)ri[b];
    float s = 0.f;
    for (int i = 0; i < npg; ++i) {
        float v = out2[(size_t)(start + i) * 128 + c];
        s += v > 0.f ? v : 0.f;
    }
    g[b * 512 + goff + c]       = s / (float)npg;
    g[b * 512 + goff + 128 + c] = rootx2[b * 128 + c];
}

// final MLP: [B,512] -> relu[256] -> [B,2]
__global__ void k_mlp(const float* __restrict__ g, const float* __restrict__ w1,
                      const float* __restrict__ b1, const float* __restrict__ w2,
                      const float* __restrict__ b2, float* __restrict__ out) {
    __shared__ float s0[256], s1[256];
    int b = blockIdx.x, j = threadIdx.x;
    const float* gb = g + b * 512;
    float acc = b1[j];
    for (int k = 0; k < 512; ++k) acc += gb[k] * w1[k * 256 + j];
    float hm = acc > 0.f ? acc : 0.f;
    s0[j] = hm * w2[j * 2 + 0];
    s1[j] = hm * w2[j * 2 + 1];
    __syncthreads();
    for (int st = 128; st > 0; st >>= 1) {
        if (j < st) { s0[j] += s0[j + st]; s1[j] += s1[j + st]; }
        __syncthreads();
    }
    if (j == 0) {
        out[b * 2 + 0] = s0[0] + b2[0];
        out[b * 2 + 1] = s1[0] + b2[1];
    }
}

extern "C" void kernel_launch(void* const* d_in, const int* in_sizes, int n_in,
                              void* d_out, int out_size, void* d_ws, size_t ws_size,
                              hipStream_t stream) {
    const float*     x       = (const float*)d_in[0];
    const float*     xda     = (const float*)d_in[1];
    const long long* ei      = (const long long*)d_in[2];
    const long long* batch   = (const long long*)d_in[3];
    const long long* rootidx = (const long long*)d_in[4];
    const float* td_w1 = (const float*)d_in[5];
    const float* td_b1 = (const float*)d_in[6];
    const float* td_w2 = (const float*)d_in[7];
    const float* td_b2 = (const float*)d_in[8];
    const float* bu_w1 = (const float*)d_in[9];
    const float* bu_b1 = (const float*)d_in[10];
    const float* bu_w2 = (const float*)d_in[11];
    const float* bu_b2 = (const float*)d_in[12];
    const float* mlp_w1 = (const float*)d_in[13];
    const float* mlp_b1 = (const float*)d_in[14];
    const float* mlp_w2 = (const float*)d_in[15];
    const float* mlp_b2 = (const float*)d_in[16];

    const int N = in_sizes[0] / 96;
    const int E = in_sizes[2] / 2;
    const int B = in_sizes[4];
    const int npg = N / B;
    const long long* e0 = ei;
    const long long* e1 = ei + E;

    // workspace carve-up
    char* wsp = (char*)d_ws;
    auto carve = [&](size_t bytes) { char* p = wsp; wsp += (bytes + 255) & ~(size_t)255; return p; };
    __bf16* x0b    = (__bf16*)carve((size_t)N * 128 * 2);
    float*  dinvTD = (float*)carve((size_t)N * 4);
    float*  dinvBU = (float*)carve((size_t)N * 4);
    __bf16* wb     = (__bf16*)carve(128 * 128 * 2);
    float*  bufA   = (float*)carve((size_t)N * 128 * 4);   // hw / hw2
    float*  bufB   = (float*)carve((size_t)N * 128 * 4);   // h1 (x2) / out2
    __bf16* h1b    = (__bf16*)carve((size_t)N * 128 * 2);
    float*  rt     = (float*)carve((size_t)B * 128 * 4);
    float*  rx     = (float*)carve((size_t)B * 128 * 4);
    float*  g      = (float*)carve((size_t)B * 512 * 4);

    const int nbNF = (N * 128 + TPB - 1) / TPB;
    const int nbN4 = (N * 32 + TPB - 1) / TPB;
    const int nbN  = (N + TPB - 1) / TPB;
    const int nbE  = (E + TPB - 1) / TPB;
    const int nbEw = (E + 7) / 8;               // one wave per edge, 8 waves/block
    const int nbW  = (128 * 128 + TPB - 1) / TPB;
    const int nbB  = (B * 128 + TPB - 1) / TPB;

    // shared prep
    k_cast_x0<<<nbNF, TPB, 0, stream>>>(x, xda, x0b, N);
    k_fill<<<nbN, TPB, 0, stream>>>(dinvTD, 1.0f, N);
    k_fill<<<nbN, TPB, 0, stream>>>(dinvBU, 1.0f, N);
    k_count_deg<<<nbE, TPB, 0, stream>>>(e0, e1, dinvTD, dinvBU, E);
    k_rsqrt<<<nbN, TPB, 0, stream>>>(dinvTD, N);
    k_rsqrt<<<nbN, TPB, 0, stream>>>(dinvBU, N);

    struct Br {
        const float *w1, *b1, *w2, *b2;
        const long long *s, *d;
        const float* dinv;
        int goff;
    };
    // g = cat(bu, td): BU at offset 0, TD at offset 256.
    // TD: src=edge[0], dst=edge[1].  BU: src=edge[1], dst=edge[0].
    Br brs[2] = {
        { bu_w1, bu_b1, bu_w2, bu_b2, e1, e0, dinvBU, 0   },
        { td_w1, td_b1, td_w2, td_b2, e0, e1, dinvTD, 256 },
    };

    for (int bi = 0; bi < 2; ++bi) {
        const Br& br = brs[bi];
        // layer 1: hw = x0 @ W1 (bf16 WMMA, pre-packed B fragments)
        k_pack_w<<<nbW, TPB, 0, stream>>>(br.w1, wb);
        k_gemm_wmma_bf16<<<N / 16, TPB, 0, stream>>>(x0b, wb, bufA);
        k_conv_init<<<nbN4, TPB, 0, stream>>>(bufA, br.dinv, br.b1, bufB, N);
        k_conv_edge<<<nbEw, TPB, 0, stream>>>(bufA, br.s, br.d, br.dinv, bufB, E);
        // bufB = x2 (pre-relu conv-1 output)
        k_gather_root<<<nbB, TPB, 0, stream>>>(bufB, rootidx, rx, B);
        k_relu_cast<<<nbN4, TPB, 0, stream>>>(bufB, h1b, N * 32);
        // layer 2: hw2 = relu(h) @ W2_top + rt[batch]  (root half is per-graph constant)
        k_root_term<<<B, 128, 0, stream>>>(x, xda, rootidx, br.w2, rt);
        k_pack_w<<<nbW, TPB, 0, stream>>>(br.w2, wb);   // top 128 rows contiguous
        k_gemm_wmma_bf16<<<N / 16, TPB, 0, stream>>>(h1b, wb, bufA);
        k_add_rt<<<nbN4, TPB, 0, stream>>>(bufA, rt, batch, N);
        k_conv_init<<<nbN4, TPB, 0, stream>>>(bufA, br.dinv, br.b2, bufB, N);
        k_conv_edge<<<nbEw, TPB, 0, stream>>>(bufA, br.s, br.d, br.dinv, bufB, E);
        // readout: relu + segment mean + root2 concat
        k_readout<<<B, 128, 0, stream>>>(bufB, rx, rootidx, g, br.goff, npg);
    }

    k_mlp<<<B, TPB, 0, stream>>>(g, mlp_w1, mlp_b1, mlp_w2, mlp_b2, (float*)d_out);
}
